// TBGCN_2954937500461
// MI455X (gfx1250) — matrix operand order repaired and checked by reading, not compiled, not run on hardware
//
#include <hip/hip_runtime.h>

// ---------------------------------------------------------------------------
// TBGCN forward for MI455X (gfx1250, wave32, WMMA).
// Shapes (from reference): B=8, T=16, N=8192, F=C=O=64, K=3.
// Memory-bound (512MB traffic @ 23.3TB/s ~ 22us); matmuls pushed through
// v_wmma_f32_16x16x32_f16; all transcendentals use the hardware v_exp_f32
// (__expf) so the pointwise tail stays branch-free and VALU-light.
// ---------------------------------------------------------------------------

typedef __attribute__((ext_vector_type(16))) _Float16 v16h;
typedef __attribute__((ext_vector_type(8)))  _Float16 v8h;
typedef __attribute__((ext_vector_type(8)))  float    v8f;

#define EPSF        1e-14f
#define SELU_SCALE  1.0507009873554805f
#define SELU_ALPHA  1.6732632423543772f

// Branch-free SELU: always evaluate the cheap hw exp, select with cndmask.
__device__ __forceinline__ float selu_f(float x) {
  const float neg = SELU_ALPHA * (__expf(fminf(x, 0.f)) - 1.f);
  return SELU_SCALE * (x > 0.f ? x : neg);
}

// ---- CDNA5 WMMA fragment builders (wave32) --------------------------------
// 16-bit A matrix 16x32 (ISA 7.12.2): lanes 0-15 row M=lane, K={kb+0..7, kb+16..23};
// lanes 16-31 row M=lane-16, K={kb+8..15, kb+24..31}.
__device__ __forceinline__ v16h load_a_frag(const _Float16* tile /*16 rows x 64*/, int kb) {
  const int lane = threadIdx.x & 31;
  const int row  = lane & 15;
  const int kh   = (lane >> 4) << 3;          // 0 or 8
  const _Float16* p = tile + row * 64 + kb + kh;
  v8h lo = *(const v8h*)(p);
  v8h hi = *(const v8h*)(p + 16);
  v16h a;
#pragma unroll
  for (int j = 0; j < 8; ++j) { a[j] = lo[j]; a[j + 8] = hi[j]; }
  return a;
}

// 16-bit B matrix 32x16: lanes 0-15 col N=lane, K={0..7, 16..23}; lanes 16-31 col N=lane-16, K={8..15, 24..31}.
// W is row-major [K][64]; pass W (+32*64) for the kb=0 (kb=32) half.
__device__ __forceinline__ v16h load_b_frag(const _Float16* __restrict__ W, int nb) {
  const int lane = threadIdx.x & 31;
  const int n  = nb + (lane & 15);
  const int k0 = (lane >> 4) << 3;            // 0 or 8
  v16h b;
#pragma unroll
  for (int j = 0; j < 16; ++j) b[j] = W[(k0 + j + ((j >> 3) << 3)) * 64 + n];
  return b;
}

// ---------------------------------------------------------------------------
// Kernel A: rotation weights Wsum = sum_O W(H), converted to f16, plus
// f16 W_lin and f32 bias, all into workspace.
// ---------------------------------------------------------------------------
__global__ __launch_bounds__(256)
void wsum_prep_kernel(const float* __restrict__ H,        // (64,3,64)
                      const float* __restrict__ W_lin,    // (64,64)
                      const float* __restrict__ b_lin,    // (64,)
                      _Float16* __restrict__ wsum_h,      // (3,64,64) f16
                      _Float16* __restrict__ wlin_h,      // (64,64) f16
                      float*    __restrict__ bias_f) {    // (64,) f32
  __shared__ float s_ch[192], s_sh[192], s_inv[192];
  const int tid = threadIdx.x;
  if (tid < 192) {                                   // tid == o*3+k
    const float* h = H + tid * 64;
    float ss = 0.f;
#pragma unroll 8
    for (int c = 1; c < 64; ++c) ss += h[c] * h[c];
    s_inv[tid] = rsqrtf(ss + EPSF);
    const float e = __expf(h[0]), ei = 1.f / e;
    s_ch[tid] = 0.5f * (e + ei);                     // cosh(theta)
    s_sh[tid] = 0.5f * (e - ei);                     // sinh(theta)
  }
  __syncthreads();
  for (int idx = tid; idx < 3 * 64 * 64; idx += blockDim.x) {
    const int k = idx >> 12;
    const int i = (idx >> 6) & 63;
    const int j = idx & 63;
    float acc = 0.f;
    for (int o = 0; o < 64; ++o) {
      const int ok = o * 3 + k;
      const float* h = H + ok * 64;
      const float inv = s_inv[ok], ch = s_ch[ok], sh = s_sh[ok];
      float w;
      if (i == 0 && j == 0)      w = ch;
      else if (i == 0)           w = sh * h[j] * inv;
      else if (j == 0)           w = sh * h[i] * inv;
      else                       w = ((i == j) ? 1.f : 0.f)
                                     - (1.f - ch) * h[i] * h[j] * inv * inv;
      acc += w;
    }
    wsum_h[idx] = (_Float16)acc;
  }
  for (int idx = tid; idx < 64 * 64; idx += blockDim.x)
    wlin_h[idx] = (_Float16)W_lin[idx];
  if (tid < 64) bias_f[tid] = b_lin[tid];
}

// ---------------------------------------------------------------------------
// Kernel B: fused  linear+selu -> exp-map -> temporal WMMA conv -> tail.
// One block = (b, 16 n-values); 4 waves, each owning a 16-channel column tile.
// ---------------------------------------------------------------------------
__global__ __launch_bounds__(128)
void tbgcn_main_kernel(const float* __restrict__ node_repr,  // (8,16,8192,64)
                       const _Float16* __restrict__ wsum_h,  // (3,64,64) f16
                       const _Float16* __restrict__ wlin_h,  // (64,64) f16
                       const float* __restrict__ bias_f,     // (64,)
                       float* __restrict__ out) {            // (8,16,8192,64)
  extern __shared__ __align__(16) char smem_raw[];
  float*    s_x32 = (float*)smem_raw;                              // 256*64 f32 (later: nu)
  _Float16* s_h16 = (_Float16*)(smem_raw + 256 * 64 * 4);          // 256*64 f16 (in, later: x)
  float*    s_bnd = (float*)(smem_raw + 256 * 64 * 4 + 256 * 64 * 2); // 32*64 f32 (x at t=0,15)

  const int tid  = threadIdx.x;
  const int lane = tid & 31;
  const int wv   = tid >> 5;          // 0..3: column tile
  const int nb   = wv << 4;           // channel base of this wave
  const int ntiles = 8192 / 16;
  const int b  = blockIdx.x / ntiles;
  const int n0 = (blockIdx.x % ntiles) << 4;

  __builtin_prefetch(wsum_h, 0, 0);   // global_prefetch_b8: warm the shared weights
  __builtin_prefetch(wlin_h, 0, 0);

  // --- per-wave B fragments (held in VGPRs for the whole kernel) -----------
  v16h bw0 = load_b_frag(wlin_h, nb);
  v16h bw1 = load_b_frag(wlin_h + 32 * 64, nb);
  v16h bk[3][2];
#pragma unroll
  for (int k = 0; k < 3; ++k) {
    bk[k][0] = load_b_frag(wsum_h + k * 4096, nb);
    bk[k][1] = load_b_frag(wsum_h + k * 4096 + 32 * 64, nb);
  }
  const float bias = bias_f[nb + (lane & 15)];

  // --- phase 1: stage input tile to LDS as f16 -----------------------------
#pragma unroll
  for (int rr = 0; rr < 2; ++rr) {
    const int ri = tid * 2 + rr;                 // 0..255 = t*16 + r
    const int t = ri >> 4, r = ri & 15;
    const float* src = node_repr + (((size_t)b * 16 + t) * 8192 + n0 + r) * 64;
    _Float16* dst = s_h16 + ri * 64;
#pragma unroll
    for (int c = 0; c < 64; c += 8) {
      float4 f0 = *(const float4*)(src + c);
      float4 f1 = *(const float4*)(src + c + 4);
      v8h h;
      h[0] = (_Float16)f0.x; h[1] = (_Float16)f0.y; h[2] = (_Float16)f0.z; h[3] = (_Float16)f0.w;
      h[4] = (_Float16)f1.x; h[5] = (_Float16)f1.y; h[6] = (_Float16)f1.z; h[7] = (_Float16)f1.w;
      *(v8h*)(dst + c) = h;
    }
  }
  __syncthreads();

  // --- phase 2: x = selu(in @ W_lin + b) via WMMA --------------------------
  const int colg  = nb + (lane & 15);
  const int rhalf = (lane >> 4) << 3;            // 0 or 8 (C/D layout row offset)
  for (int t = 0; t < 16; ++t) {
    v16h a0 = load_a_frag(s_h16 + t * 16 * 64, 0);
    v16h a1 = load_a_frag(s_h16 + t * 16 * 64, 32);
    v8f acc = {};
    acc = __builtin_amdgcn_wmma_f32_16x16x32_f16(false, a0, false, bw0, (short)0, acc, false, false);
    acc = __builtin_amdgcn_wmma_f32_16x16x32_f16(false, a1, false, bw1, (short)0, acc, false, false);
#pragma unroll
    for (int r = 0; r < 8; ++r) {
      const float z = selu_f(acc[r] + bias);
      s_x32[(t * 16 + rhalf + r) * 64 + colg] = z;
    }
  }
  __syncthreads();

  // --- phase 3: exp-map (per row), rewrite LDS as f16 x; keep t=0,15 in f32 -
#pragma unroll
  for (int rr = 0; rr < 2; ++rr) {
    const int ri = tid + rr * 128;               // 0..255
    const float* src = s_x32 + ri * 64;
    _Float16* dst = s_h16 + ri * 64;
    float ss = 0.f;
#pragma unroll 8
    for (int c = 1; c < 64; ++c) { const float v = src[c]; ss += v * v; }
    const float nrm = sqrtf(ss + EPSF);
    const float en = __expf(nrm), eni = 1.f / en;
    const float x0 = 0.5f * (en + eni);          // cosh
    const float sc = 0.5f * (en - eni) / nrm;    // sinh/nrm
    const int t = ri >> 4;
    float* bdst = (t == 0)  ? (s_bnd + (ri & 15) * 64)
                : (t == 15) ? (s_bnd + (16 + (ri & 15)) * 64) : nullptr;
    dst[0] = (_Float16)x0;
    if (bdst) bdst[0] = x0;
#pragma unroll 8
    for (int c = 1; c < 64; ++c) {
      const float xs = sc * src[c];
      dst[c] = (_Float16)xs;
      if (bdst) bdst[c] = xs;
    }
  }
  __syncthreads();

  // --- phase 4: temporal conv nu[t] = sum_k x[t+k] @ Wsum[k] (WMMA) --------
  float* s_nu = s_x32;                           // alias: x32 staging is dead
  for (int t = 0; t < 14; ++t) {
    v8f acc = {};
#pragma unroll
    for (int k = 0; k < 3; ++k) {
      v16h a0 = load_a_frag(s_h16 + (t + k) * 16 * 64, 0);
      v16h a1 = load_a_frag(s_h16 + (t + k) * 16 * 64, 32);
      acc = __builtin_amdgcn_wmma_f32_16x16x32_f16(false, a0, false, bk[k][0], (short)0, acc, false, false);
      acc = __builtin_amdgcn_wmma_f32_16x16x32_f16(false, a1, false, bk[k][1], (short)0, acc, false, false);
    }
#pragma unroll
    for (int r = 0; r < 8; ++r)
      s_nu[(t * 16 + rhalf + r) * 64 + colg] = acc[r] * (1.0f / 192.0f);
  }
  __syncthreads();

  // --- phase 5: per-row tail (l_dot/coef + Klein projection + clip) --------
#pragma unroll
  for (int rr = 0; rr < 2; ++rr) {
    const int ri = tid + rr * 128;               // 0..255: 224 window rows + 32 boundary rows
    const float* src;
    float coef;
    int time, r;
    if (ri < 224) {
      const int t = ri >> 4; r = ri & 15; time = t + 1;
      src = s_nu + ri * 64;
      const float nu0 = src[0];
      float ss = 0.f;
#pragma unroll 8
      for (int c = 0; c < 64; ++c) { const float v = src[c]; ss += v * v; }
      coef = rsqrtf(fabsf(ss - 2.f * nu0 * nu0));
    } else {
      const int q = ri - 224;                    // 0..31
      time = (q < 16) ? 0 : 15;
      r = q & 15;
      src = s_bnd + q * 64;
      coef = 1.f;
    }
    const float o0   = coef * src[0];
    const float inv0 = 1.f / (o0 + 1.f);         // sc = sqrt(C_CURV) = 1
    float nsq = 0.f;
#pragma unroll 8
    for (int c = 1; c < 64; ++c) {
      const float p = selu_f(coef * src[c] * inv0);
      nsq += p * p;
    }
    const float denom = 1.f - nsq + EPSF;
    const float t2    = 2.f / denom;
    const float sws2  = nsq * t2 * t2;           // sum(ws^2) pre-clip
    const float wn    = sqrtf(sws2 + EPSF);
    const float clip  = fminf(1.f, 10000.f / wn);
    const float psc   = t2 * clip;
    const float first = sqrtf(1.f + sws2 * clip * clip);
    float* dst = out + (((size_t)b * 16 + time) * 8192 + n0 + r) * 64;
    float4 v0;
    v0.x = first;
    v0.y = selu_f(coef * src[1] * inv0) * psc;
    v0.z = selu_f(coef * src[2] * inv0) * psc;
    v0.w = selu_f(coef * src[3] * inv0) * psc;
    *(float4*)dst = v0;
#pragma unroll
    for (int c = 4; c < 64; c += 4) {
      float4 u;
      u.x = selu_f(coef * src[c + 0] * inv0) * psc;
      u.y = selu_f(coef * src[c + 1] * inv0) * psc;
      u.z = selu_f(coef * src[c + 2] * inv0) * psc;
      u.w = selu_f(coef * src[c + 3] * inv0) * psc;
      *(float4*)(dst + c) = u;
    }
  }
}

// ---------------------------------------------------------------------------
extern "C" void kernel_launch(void* const* d_in, const int* in_sizes, int n_in,
                              void* d_out, int out_size, void* d_ws, size_t ws_size,
                              hipStream_t stream) {
  const float* node_repr = (const float*)d_in[0];   // (8,16,8192,64)
  const float* H         = (const float*)d_in[1];   // (64,3,64)
  const float* W_lin     = (const float*)d_in[2];   // (64,64)
  const float* b_lin     = (const float*)d_in[3];   // (64,)

  _Float16* wsum_h = (_Float16*)d_ws;                               // 24576 B
  _Float16* wlin_h = (_Float16*)((char*)d_ws + 3 * 64 * 64 * 2);    //  8192 B
  float*    bias_f = (float*)((char*)d_ws + 3 * 64 * 64 * 2 + 64 * 64 * 2); // 256 B

  wsum_prep_kernel<<<1, 256, 0, stream>>>(H, W_lin, b_lin, wsum_h, wlin_h, bias_f);

  const size_t lds_bytes = 256 * 64 * 4   // s_x32 / s_nu
                         + 256 * 64 * 2   // s_h16 (input f16 / x f16)
                         + 32 * 64 * 4;   // s_bnd (x at t=0,15)
  const dim3 grid(8 * (8192 / 16));       // 4096 blocks: (b, n-tile)
  tbgcn_main_kernel<<<grid, 128, lds_bytes, stream>>>(node_repr, wsum_h, wlin_h,
                                                      bias_f, (float*)d_out);
}
